// GlobalPMFSBlock_AP_21371757265548
// MI455X (gfx1250) — matrix operand order repaired and checked by reading, not compiled, not run on hardware
//
#include <hip/hip_runtime.h>
#include <stdint.h>

// ---------------- types ----------------
typedef __bf16 bf16;
typedef __attribute__((ext_vector_type(8)))  __bf16 v8bf;
typedef __attribute__((ext_vector_type(16))) __bf16 v16bf;
typedef __attribute__((ext_vector_type(8)))  float  v8f;

__device__ __forceinline__ float bf2f(bf16 x) {
    unsigned short u = __builtin_bit_cast(unsigned short, x);
    unsigned int v = ((unsigned int)u) << 16;
    return __builtin_bit_cast(float, v);
}
__device__ __forceinline__ bf16 f2bf(float f) {
    unsigned int u = __builtin_bit_cast(unsigned int, f);
    unsigned int lsb = (u >> 16) & 1u;
    u += 0x7fffu + lsb;                       // round-to-nearest-even
    unsigned short h = (unsigned short)(u >> 16);
    return __builtin_bit_cast(bf16, h);
}
__device__ __forceinline__ v8f wmma_bf16(v16bf a, v16bf b, v8f c) {
    // D = A(16x32 bf16) * B(32x16 bf16) + C(16x16 f32)
    return __builtin_amdgcn_wmma_f32_16x16x32_bf16(false, a, false, b, (short)0, c, false, false);
}

#define NPIX 16384        // 16 images * 32*32
#define CHIN 192

union ABfrag { v16bf v; v8bf h[2]; };

// ---------------- stage 0: maxpool NCHW f32 -> NHWC bf16 ----------------
__global__ void pool_kernel(const float* __restrict__ in, bf16* __restrict__ out,
                            int C, int Hin, int P) {
    long idx = (long)blockIdx.x * blockDim.x + threadIdx.x;
    long total = 16L * C * 32 * 32;
    if (idx >= total) return;
    int ox = (int)(idx & 31); long t = idx >> 5;
    int oy = (int)(t & 31);   t >>= 5;
    int c  = (int)(t % C);
    int b  = (int)(t / C);
    const float* base = in + (((long)(b * C + c) * Hin + (long)oy * P) * Hin + (long)ox * P);
    float m = -__builtin_inff();
    for (int dy = 0; dy < P; ++dy) {
        const float* row = base + (long)dy * Hin;
        for (int dx = 0; dx < P; ++dx) m = fmaxf(m, row[dx]);
    }
    out[((long)(b * 1024 + oy * 32 + ox)) * C + c] = f2bf(m);
}

// -------- weight transform: OIHW f32 -> [cout][(ky*K+kx)*Cin+ci] bf16, BN folded --------
__global__ void prep_weights(const float* __restrict__ w, const float* __restrict__ g,
                             const float* __restrict__ bb, const float* __restrict__ m,
                             const float* __restrict__ v, bf16* __restrict__ wt,
                             float* __restrict__ bias, int Cout, int Cin, int K) {
    int KK = K * K * Cin;
    long idx = (long)blockIdx.x * blockDim.x + threadIdx.x;
    if (idx >= (long)Cout * KK) return;
    int co  = (int)(idx / KK);
    int rem = (int)(idx % KK);
    int kk  = rem / Cin, ci = rem % Cin;
    int ky  = kk / K,    kx = kk % K;
    float inv = g[co] * rsqrtf(v[co] + 1e-5f);
    float val = w[(((long)co * Cin + ci) * K + ky) * K + kx] * inv;
    wt[(long)co * KK + rem] = f2bf(val);
    if (rem == 0) bias[co] = bb[co] - m[co] * inv;
}

// ---------------- implicit-GEMM 3x3 conv (pad 1), M32xN64 per wave ----------------
// in: bf16 NHWC [NPIX][CIN]; wt: [Cout][9*CIN]; bias f32[Cout]
// outMode 0: bf16 out[pix*192 + coff + cout]    outMode 1: f32 NCHW d_out
template<int CIN>
__global__ __launch_bounds__(128)
void conv3x3_wmma(const bf16* __restrict__ in, const bf16* __restrict__ wt,
                  const float* __restrict__ bias, bf16* __restrict__ outB,
                  float* __restrict__ outF, int Cout, int coff, int outMode) {
    constexpr int NC = CIN / 32;
    constexpr int KK = 9 * CIN;
    const int lane = threadIdx.x & 31;
    const int wave = threadIdx.x >> 5;
    const int m0   = blockIdx.x * 128 + wave * 32;  // wave = one full image row (32 px)
    const int n0   = blockIdx.y * 64;
    const int img  = m0 >> 10;
    const int py   = (m0 & 1023) >> 5;
    const int lm   = lane & 15;
    const int hiK  = (lane >= 16);

    v8f acc[2][4] = {};
    for (int ky = 0; ky < 3; ++ky) {
        int iy = py + ky - 1;
        if (iy < 0 || iy > 31) continue;            // zero row contributes nothing
        const bf16* rowbase = in + ((long)(img * 1024 + iy * 32)) * CIN;
        for (int kx = 0; kx < 3; ++kx) {
            int ix0 = lm + kx - 1;                  // A row group 0: px 0..15
            int ix1 = ix0 + 16;                     // A row group 1: px 16..31
            bool in0 = (ix0 >= 0);
            bool in1 = (ix1 <= 31);
            const bf16* a0 = rowbase + (long)ix0 * CIN;
            const bf16* a1 = rowbase + (long)ix1 * CIN;
            int kbase = (ky * 3 + kx) * CIN;
#pragma unroll
            for (int cc = 0; cc < NC; ++cc) {
                int c0 = cc * 32 + (hiK ? 8 : 0);
                ABfrag A0, A1;
                if (in0) { A0.h[0] = *(const v8bf*)(a0 + c0); A0.h[1] = *(const v8bf*)(a0 + c0 + 16); }
                else     { for (int i = 0; i < 16; ++i) A0.v[i] = f2bf(0.f); }
                if (in1) { A1.h[0] = *(const v8bf*)(a1 + c0); A1.h[1] = *(const v8bf*)(a1 + c0 + 16); }
                else     { for (int i = 0; i < 16; ++i) A1.v[i] = f2bf(0.f); }
                int kb = kbase + cc * 32 + (hiK ? 8 : 0);
                ABfrag Bf[4];                        // preload all B frags (one clause, no WAR)
#pragma unroll
                for (int nb = 0; nb < 4; ++nb) {
                    const bf16* brow = wt + (long)(n0 + nb * 16 + lm) * KK + kb;
                    Bf[nb].h[0] = *(const v8bf*)(brow);
                    Bf[nb].h[1] = *(const v8bf*)(brow + 16);
                }
#pragma unroll
                for (int nb = 0; nb < 4; ++nb) {
                    acc[0][nb] = wmma_bf16(A0.v, Bf[nb].v, acc[0][nb]);   // B reused twice
                    acc[1][nb] = wmma_bf16(A1.v, Bf[nb].v, acc[1][nb]);
                }
            }
        }
    }
    for (int g = 0; g < 2; ++g)
        for (int nb = 0; nb < 4; ++nb) {
            int cout = n0 + nb * 16 + lm;
            float bi = bias[cout];
            for (int e = 0; e < 8; ++e) {
                int pix = m0 + g * 16 + (hiK ? 8 + e : e);
                float d = fmaxf(acc[g][nb][e] + bi, 0.f);
                if (outMode == 0) outB[(long)pix * CHIN + coff + cout] = f2bf(d);
                else outF[(((long)(img * Cout + cout)) << 10) + (pix & 1023)] = d;
            }
        }
}

// ---------------- 1x1 conv = GEMM [NPIX x 192] * [192 x 192], M32xN64/wave ----------------
__global__ __launch_bounds__(128)
void gemm1x1_wmma(const bf16* __restrict__ in, const bf16* __restrict__ wt,
                  const float* __restrict__ bias, bf16* __restrict__ out) {
    constexpr int NC = CHIN / 32;   // 6, fully unrolled
    const int lane = threadIdx.x & 31;
    const int wave = threadIdx.x >> 5;
    const int m0   = blockIdx.x * 128 + wave * 32;
    const int n0   = blockIdx.y * 64;
    const int lm   = lane & 15;
    const int hiK  = (lane >= 16);
    const bf16* arow0 = in + ((long)(m0 + lm)) * CHIN;
    const bf16* arow1 = in + ((long)(m0 + 16 + lm)) * CHIN;

    v8f acc[2][4] = {};
#pragma unroll
    for (int cc = 0; cc < NC; ++cc) {
        int c0 = cc * 32 + (hiK ? 8 : 0);
        ABfrag A0, A1;
        A0.h[0] = *(const v8bf*)(arow0 + c0); A0.h[1] = *(const v8bf*)(arow0 + c0 + 16);
        A1.h[0] = *(const v8bf*)(arow1 + c0); A1.h[1] = *(const v8bf*)(arow1 + c0 + 16);
        int kb = cc * 32 + (hiK ? 8 : 0);
        ABfrag Bf[4];                                // preload all B frags
#pragma unroll
        for (int nb = 0; nb < 4; ++nb) {
            const bf16* brow = wt + (long)(n0 + nb * 16 + lm) * CHIN + kb;
            Bf[nb].h[0] = *(const v8bf*)(brow);
            Bf[nb].h[1] = *(const v8bf*)(brow + 16);
        }
#pragma unroll
        for (int nb = 0; nb < 4; ++nb) {
            acc[0][nb] = wmma_bf16(A0.v, Bf[nb].v, acc[0][nb]);
            acc[1][nb] = wmma_bf16(A1.v, Bf[nb].v, acc[1][nb]);
        }
    }
    for (int g = 0; g < 2; ++g)
        for (int nb = 0; nb < 4; ++nb) {
            int cout = n0 + nb * 16 + lm;
            float bi = bias[cout];
            for (int e = 0; e < 8; ++e) {
                int pix = m0 + g * 16 + (hiK ? 8 + e : e);
                out[(long)pix * CHIN + cout] = f2bf(fmaxf(acc[g][nb][e] + bi, 0.f));
            }
        }
}

// ---------------- ch_K: 192 -> 1 conv (wave per pixel) ----------------
__global__ void chk_kernel(const bf16* __restrict__ x, const bf16* __restrict__ wk,
                           const float* __restrict__ bias, float* __restrict__ out) {
    int wid  = (blockIdx.x * blockDim.x + threadIdx.x) >> 5;
    int lane = threadIdx.x & 31;
    if (wid >= NPIX) return;
    const bf16* row = x + (long)wid * CHIN;
    float s = 0.f;
    for (int i = 0; i < 6; ++i) { int c = lane + i * 32; s += bf2f(row[c]) * bf2f(wk[c]); }
    for (int off = 16; off > 0; off >>= 1) s += __shfl_xor(s, off, 32);
    if (lane == 0) out[wid] = fmaxf(s + bias[0], 0.f);
}

// ---------------- softmax over HW (1024) per batch, in-place ----------------
__global__ void softmax_hw(float* __restrict__ k) {
    __shared__ float red[256];
    int b = blockIdx.x, t = threadIdx.x;
    float* p = k + b * 1024;
    float vals[4]; float mx = -__builtin_inff();
    for (int i = 0; i < 4; ++i) { vals[i] = p[t + i * 256]; mx = fmaxf(mx, vals[i]); }
    red[t] = mx; __syncthreads();
    for (int off = 128; off > 0; off >>= 1) { if (t < off) red[t] = fmaxf(red[t], red[t + off]); __syncthreads(); }
    mx = red[0]; __syncthreads();
    float s = 0.f;
    for (int i = 0; i < 4; ++i) { vals[i] = __expf(vals[i] - mx); s += vals[i]; }
    red[t] = s; __syncthreads();
    for (int off = 128; off > 0; off >>= 1) { if (t < off) red[t] += red[t + off]; __syncthreads(); }
    float inv = 1.f / red[0];
    for (int i = 0; i < 4; ++i) p[t + i * 256] = vals[i] * inv;
}

// ---------------- Z[b][c] = sum_n Q[b,n,c] * K[b,n] ----------------
__global__ void channel_Z(const bf16* __restrict__ q, const float* __restrict__ k,
                          float* __restrict__ Z) {
    int b = blockIdx.x, c = threadIdx.x;       // 192 threads
    const bf16*  qb = q + (long)b * 1024 * CHIN;
    const float* kb = k + b * 1024;
    float s = 0.f;
    for (int n = 0; n < 1024; ++n) s += bf2f(qb[(long)n * CHIN + c]) * kb[n];
    Z[b * CHIN + c] = s;
}

// ---------------- score GEMM (16x192x192) + LayerNorm + sigmoid ----------------
__global__ void score_ln(const float* __restrict__ Z, const float* __restrict__ sw,
                         const float* __restrict__ sb, const float* __restrict__ lg,
                         const float* __restrict__ lb, float* __restrict__ score) {
    __shared__ float zs[CHIN];
    __shared__ float red[CHIN];
    int b = blockIdx.x, c = threadIdx.x;
    zs[c] = Z[b * CHIN + c]; __syncthreads();
    float a = sb[c];
    for (int j = 0; j < CHIN; ++j) a += zs[j] * sw[c * CHIN + j];
    red[c] = a; __syncthreads();
    for (int off = 128; off > 0; off >>= 1) { if (c < off && c + off < CHIN) red[c] += red[c + off]; __syncthreads(); }
    float mu = red[0] * (1.f / CHIN); __syncthreads();
    float d = a - mu;
    red[c] = d * d; __syncthreads();
    for (int off = 128; off > 0; off >>= 1) { if (c < off && c + off < CHIN) red[c] += red[c + off]; __syncthreads(); }
    float var = red[0] * (1.f / CHIN);
    float t = d * rsqrtf(var + 1e-5f) * lg[c] + lb[c];
    score[b * CHIN + c] = 1.f / (1.f + __expf(-t));
}

// ---------------- ch_out = ch_V * sigmoid-score ----------------
__global__ void apply_ch_score(const bf16* __restrict__ v, const float* __restrict__ score,
                               bf16* __restrict__ out) {
    long idx = (long)blockIdx.x * blockDim.x + threadIdx.x;
    if (idx >= (long)NPIX * CHIN) return;
    int c = (int)(idx % CHIN);
    int b = (int)(idx / ((long)CHIN * 1024));
    out[idx] = f2bf(bf2f(v[idx]) * score[b * CHIN + c]);
}

// ---------------- sp_K: mean over (h,w,br) then softmax over 64 channels ----------------
__global__ void spk_reduce(const bf16* __restrict__ spk, float* __restrict__ softK) {
    __shared__ float red[256];
    __shared__ float sm[64];
    int b = blockIdx.x, t = threadIdx.x;
    int ck = t & 63, part = t >> 6;            // 4 partial sums per channel
    const bf16* base = spk + (long)b * 1024 * CHIN;
    float s = 0.f;
    for (int n = part * 256; n < part * 256 + 256; ++n) {
        const bf16* row = base + (long)n * CHIN;
        s += bf2f(row[ck]) + bf2f(row[64 + ck]) + bf2f(row[128 + ck]);
    }
    red[t] = s; __syncthreads();
    if (part == 0) sm[ck] = (red[ck] + red[ck + 64] + red[ck + 128] + red[ck + 192]) * (1.f / 3072.f);
    __syncthreads();
    if (t == 0) {
        float mx = -__builtin_inff();
        for (int i = 0; i < 64; ++i) mx = fmaxf(mx, sm[i]);
        float ss = 0.f;
        for (int i = 0; i < 64; ++i) { sm[i] = __expf(sm[i] - mx); ss += sm[i]; }
        float inv = 1.f / ss;
        for (int i = 0; i < 64; ++i) softK[b * 64 + i] = sm[i] * inv;
    }
}

// ---------------- sp_out = sp_V * sigmoid(softK . sp_Q)  (wave per pixel, in-place on V) --
__global__ void spatial_apply(const bf16* __restrict__ spq, bf16* __restrict__ spv,
                              const float* __restrict__ softK) {
    int wid  = (blockIdx.x * blockDim.x + threadIdx.x) >> 5;
    int lane = threadIdx.x & 31;
    if (wid >= NPIX) return;
    int b = wid >> 10;
    float sk0 = softK[b * 64 + lane];
    float sk1 = softK[b * 64 + 32 + lane];
    const bf16* qrow = spq + (long)wid * CHIN;
    bf16*       vrow = spv + (long)wid * CHIN;
    for (int r = 0; r < 3; ++r) {
        float pz = sk0 * bf2f(qrow[r * 64 + lane]) + sk1 * bf2f(qrow[r * 64 + 32 + lane]);
        for (int off = 16; off > 0; off >>= 1) pz += __shfl_xor(pz, off, 32);
        float s = 1.f / (1.f + __expf(-pz));
        vrow[r * 64 + lane]      = f2bf(bf2f(vrow[r * 64 + lane]) * s);
        vrow[r * 64 + 32 + lane] = f2bf(bf2f(vrow[r * 64 + 32 + lane]) * s);
    }
}

// =====================================================================
extern "C" void kernel_launch(void* const* d_in, const int* in_sizes, int n_in,
                              void* d_out, int out_size, void* d_ws, size_t ws_size,
                              hipStream_t stream) {
    (void)in_sizes; (void)n_in; (void)out_size; (void)ws_size;
    int i = 0;
    const float* fm0 = (const float*)d_in[i++];
    const float* fm1 = (const float*)d_in[i++];
    const float* fm2 = (const float*)d_in[i++];
    struct CBR { const float *w, *g, *b, *m, *v; };
    auto getCBR = [&]() { CBR p; p.w = (const float*)d_in[i++]; p.g = (const float*)d_in[i++];
                          p.b = (const float*)d_in[i++]; p.m = (const float*)d_in[i++];
                          p.v = (const float*)d_in[i++]; return p; };
    CBR cc0 = getCBR(), cc1 = getCBR(), cc2 = getCBR();
    CBR chq = getCBR(), chk = getCBR(), chv = getCBR();
    const float* score_w = (const float*)d_in[i++];
    const float* score_b = (const float*)d_in[i++];
    const float* ln_g    = (const float*)d_in[i++];
    const float* ln_b    = (const float*)d_in[i++];
    CBR spq = getCBR(), spk = getCBR(), spv = getCBR(), spo = getCBR(), outc = getCBR();
    float* out = (float*)d_out;

    // -------- bump allocator over d_ws (~42 MB total) --------
    uint8_t* base = (uint8_t*)d_ws;
    size_t cur = 0;
    auto alloc = [&](size_t bytes) -> void* {
        void* p = base + cur;
        cur = (cur + bytes + 255) & ~(size_t)255;
        return p;
    };
    bf16* pooled0 = (bf16*)alloc((size_t)NPIX *  64 * 2);
    bf16* pooled1 = (bf16*)alloc((size_t)NPIX * 128 * 2);
    bf16* pooled2 = (bf16*)alloc((size_t)NPIX * 256 * 2);  // reused later for spk output
    bf16* xact    = (bf16*)alloc((size_t)NPIX * CHIN * 2);
    bf16* bufA    = (bf16*)alloc((size_t)NPIX * CHIN * 2); // chq out -> spq out
    bf16* bufB    = (bf16*)alloc((size_t)NPIX * CHIN * 2); // chv out -> spv out -> sp_out
    bf16* bufC    = (bf16*)alloc((size_t)NPIX * CHIN * 2); // ch_out  -> spo out
    float* chkbuf = (float*)alloc((size_t)NPIX * 4);
    float* Zbuf   = (float*)alloc(16 * CHIN * 4);
    float* scbuf  = (float*)alloc(16 * CHIN * 4);
    float* softK  = (float*)alloc(16 * 64 * 4);
    bf16* wcc0 = (bf16*)alloc( 64 *  576 * 2); float* bcc0 = (float*)alloc( 64 * 4);
    bf16* wcc1 = (bf16*)alloc( 64 * 1152 * 2); float* bcc1 = (float*)alloc( 64 * 4);
    bf16* wcc2 = (bf16*)alloc( 64 * 2304 * 2); float* bcc2 = (float*)alloc( 64 * 4);
    bf16* wchq = (bf16*)alloc(192 *  192 * 2); float* bchq = (float*)alloc(192 * 4);
    bf16* wchk = (bf16*)alloc(       192 * 2); float* bchk = (float*)alloc(      4);
    bf16* wchv = (bf16*)alloc(192 *  192 * 2); float* bchv = (float*)alloc(192 * 4);
    bf16* wspq = (bf16*)alloc(192 *  192 * 2); float* bspq = (float*)alloc(192 * 4);
    bf16* wspk = (bf16*)alloc(192 *  192 * 2); float* bspk = (float*)alloc(192 * 4);
    bf16* wspv = (bf16*)alloc(192 *  192 * 2); float* bspv = (float*)alloc(192 * 4);
    bf16* wspo = (bf16*)alloc(192 *  192 * 2); float* bspo = (float*)alloc(192 * 4);
    bf16* wout = (bf16*)alloc(256 * 1728 * 2); float* bout = (float*)alloc(256 * 4);
    bf16* spkbuf = pooled2;   // pooling consumed by then; 8 MB >= 6 MB

    auto prep = [&](const CBR& p, bf16* wt, float* bs, int Cout, int Cin, int K) {
        long tot = (long)Cout * K * K * Cin;
        prep_weights<<<dim3((unsigned)((tot + 255) / 256)), 256, 0, stream>>>(
            p.w, p.g, p.b, p.m, p.v, wt, bs, Cout, Cin, K);
    };
    prep(cc0, wcc0, bcc0, 64, 64, 3);
    prep(cc1, wcc1, bcc1, 64, 128, 3);
    prep(cc2, wcc2, bcc2, 64, 256, 3);
    prep(chq, wchq, bchq, 192, 192, 1);
    prep(chk, wchk, bchk,   1, 192, 1);
    prep(chv, wchv, bchv, 192, 192, 1);
    prep(spq, wspq, bspq, 192, 192, 1);
    prep(spk, wspk, bspk, 192, 192, 1);
    prep(spv, wspv, bspv, 192, 192, 1);
    prep(spo, wspo, bspo, 192, 192, 1);
    prep(outc, wout, bout, 256, 192, 3);

    // -------- pooling (HBM-bound stage: ~470 MB streamed) --------
    pool_kernel<<<dim3(16L *  64 * 1024 / 256), 256, 0, stream>>>(fm0, pooled0,  64, 256, 8);
    pool_kernel<<<dim3(16L * 128 * 1024 / 256), 256, 0, stream>>>(fm1, pooled1, 128, 128, 4);
    pool_kernel<<<dim3(16L * 256 * 1024 / 256), 256, 0, stream>>>(fm2, pooled2, 256,  64, 2);

    // -------- branch convs -> x (concat via channel offset) --------
    conv3x3_wmma< 64><<<dim3(NPIX / 128, 1), 128, 0, stream>>>(pooled0, wcc0, bcc0, xact, nullptr, 64,   0, 0);
    conv3x3_wmma<128><<<dim3(NPIX / 128, 1), 128, 0, stream>>>(pooled1, wcc1, bcc1, xact, nullptr, 64,  64, 0);
    conv3x3_wmma<256><<<dim3(NPIX / 128, 1), 128, 0, stream>>>(pooled2, wcc2, bcc2, xact, nullptr, 64, 128, 0);

    // -------- channel attention --------
    gemm1x1_wmma<<<dim3(NPIX / 128, 3), 128, 0, stream>>>(xact, wchq, bchq, bufA);
    gemm1x1_wmma<<<dim3(NPIX / 128, 3), 128, 0, stream>>>(xact, wchv, bchv, bufB);
    chk_kernel<<<dim3(NPIX * 32 / 256), 256, 0, stream>>>(xact, wchk, bchk, chkbuf);
    softmax_hw<<<16, 256, 0, stream>>>(chkbuf);
    channel_Z<<<16, CHIN, 0, stream>>>(bufA, chkbuf, Zbuf);
    score_ln<<<16, CHIN, 0, stream>>>(Zbuf, score_w, score_b, ln_g, ln_b, scbuf);
    apply_ch_score<<<dim3((unsigned)((long)NPIX * CHIN / 256)), 256, 0, stream>>>(bufB, scbuf, bufC);

    // -------- spatial attention --------
    gemm1x1_wmma<<<dim3(NPIX / 128, 3), 128, 0, stream>>>(bufC, wspq, bspq, bufA);
    gemm1x1_wmma<<<dim3(NPIX / 128, 3), 128, 0, stream>>>(bufC, wspk, bspk, spkbuf);
    gemm1x1_wmma<<<dim3(NPIX / 128, 3), 128, 0, stream>>>(bufC, wspv, bspv, bufB);
    spk_reduce<<<16, 256, 0, stream>>>(spkbuf, softK);
    spatial_apply<<<dim3(NPIX * 32 / 256), 256, 0, stream>>>(bufA, bufB, softK);  // bufB := sp_out

    // -------- output convs --------
    gemm1x1_wmma<<<dim3(NPIX / 128, 3), 128, 0, stream>>>(bufB, wspo, bspo, bufC);
    conv3x3_wmma<192><<<dim3(NPIX / 128, 4), 128, 0, stream>>>(bufC, wout, bout, nullptr, out, 256, 0, 1);
}
